// MoELayerDistributedOptimized_67903432949842
// MI455X (gfx1250) — compile-verified, matrix operand order checked by reading, not run on hardware
//
#include <hip/hip_runtime.h>
#include <hip/hip_bf16.h>
#include <math.h>

#define N_EXP   32
#define DMODEL  1024
#define DFF     4096
#define S_TOK   8192
#define CAP     320   // ceil(8192/32 * 1.25)

#define ASTRIDE 36    // LDS row stride in ushorts: 72B rows -> b64-aligned, even k stays dword-aligned

typedef __attribute__((ext_vector_type(16))) __bf16 v16bf;
typedef __attribute__((ext_vector_type(8)))  float  v8f;

__device__ __forceinline__ unsigned short f2bf(float f) {
    union { float f; unsigned u; } v; v.f = f;
    unsigned r = v.u + 0x7fffu + ((v.u >> 16) & 1u);   // round-to-nearest-even
    return (unsigned short)(r >> 16);
}

union FragAB { unsigned u[8]; v16bf v; };

// ---------------- init ----------------
__global__ void k_init(float* imp, float* loadc, int* slot_pair) {
    int i = blockIdx.x * blockDim.x + threadIdx.x;
    if (i < N_EXP) { imp[i] = 0.f; loadc[i] = 0.f; }
    if (i < N_EXP * CAP) slot_pair[i] = -1;
}

// ---------------- router: one wave32 per token, lane == expert ----------------
__global__ void k_router(const float* __restrict__ x, const float* __restrict__ wg,
                         int* __restrict__ expert_idx, float* __restrict__ gate,
                         float* imp, float* loadc) {
    __shared__ float s_imp[N_EXP];
    __shared__ float s_load[N_EXP];
    int tid = threadIdx.x;
    if (tid < N_EXP) { s_imp[tid] = 0.f; s_load[tid] = 0.f; }
    __syncthreads();

    int wave = tid >> 5, lane = tid & 31;
    int t = blockIdx.x * 8 + wave;

    const float4* xr = (const float4*)(x + (size_t)t * DMODEL);
    const float4* wr = (const float4*)(wg + (size_t)lane * DMODEL);
    float acc = 0.f;
    for (int d = 0; d < DMODEL / 4; ++d) {
        float4 a = xr[d], b = wr[d];
        acc += a.x * b.x + a.y * b.y + a.z * b.z + a.w * b.w;
    }
    // wave softmax over 32 experts
    float m = acc;
    for (int off = 16; off > 0; off >>= 1) m = fmaxf(m, __shfl_xor(m, off, 32));
    float p = expf(acc - m);
    float sum = p;
    for (int off = 16; off > 0; off >>= 1) sum += __shfl_xor(sum, off, 32);
    float prob = p / sum;
    atomicAdd(&s_imp[lane], prob);

    // top-1 (tie -> lower expert index, matching lax.top_k)
    float v0 = prob; int e0 = lane;
    for (int off = 16; off > 0; off >>= 1) {
        float ov = __shfl_xor(v0, off, 32); int oe = __shfl_xor(e0, off, 32);
        if (ov > v0 || (ov == v0 && oe < e0)) { v0 = ov; e0 = oe; }
    }
    // top-2
    float v1 = (lane == e0) ? -1e30f : prob; int e1 = lane;
    for (int off = 16; off > 0; off >>= 1) {
        float ov = __shfl_xor(v1, off, 32); int oe = __shfl_xor(e1, off, 32);
        if (ov > v1 || (ov == v1 && oe < e1)) { v1 = ov; e1 = oe; }
    }
    if (lane == 0) {
        expert_idx[2 * t + 0] = e0; gate[2 * t + 0] = v0;
        expert_idx[2 * t + 1] = e1; gate[2 * t + 1] = v1;
        atomicAdd(&s_load[e0], 1.f);
        atomicAdd(&s_load[e1], 1.f);
    }
    __syncthreads();
    if (tid < N_EXP) {
        atomicAdd(&imp[tid], s_imp[tid]);
        atomicAdd(&loadc[tid], s_load[tid]);
    }
}

// ---------------- ordered prefix scan: positions per expert (stable-sort semantics) ----------------
__global__ void k_scan(const int* __restrict__ expert_idx, int* __restrict__ pair_pos,
                       int* __restrict__ slot_pair) {
    int e = blockIdx.x;
    int lane = threadIdx.x;
    int count = 0;
    int maxOver = -1;   // last pair writing the clamped slot wins (scatter last-write)
    for (int base = 0; base < S_TOK * 2; base += 32) {
        int p = base + lane;
        bool match = (expert_idx[p] == e);
        unsigned long long mask = __ballot(match);
        int prefix = __popcll(mask & ((1ull << lane) - 1ull));
        if (match) {
            int pos = count + prefix;
            int posc = min(pos, CAP - 1);
            pair_pos[p] = posc;
            if (pos < CAP - 1) slot_pair[e * CAP + posc] = p;
            else               maxOver = p;
        }
        count += __popcll(mask);
    }
    for (int off = 16; off > 0; off >>= 1) maxOver = max(maxOver, __shfl_xor(maxOver, off, 32));
    if (lane == 0 && maxOver >= 0) slot_pair[e * CAP + CAP - 1] = maxOver;
}

// ---------------- dispatch: gather tokens into bf16 expert_input [E,CAP,D] ----------------
__global__ void k_dispatch(const float* __restrict__ x, const int* __restrict__ slot_pair,
                           unsigned short* __restrict__ Xb) {
    int slot = blockIdx.x;
    int sp = slot_pair[slot];
    unsigned short* dst = Xb + (size_t)slot * DMODEL;
    if (sp < 0) {
        for (int i = threadIdx.x; i < DMODEL; i += blockDim.x) dst[i] = 0;
    } else {
        int t = sp >> 1;  // pair index -> token (K==2)
        const float* src = x + (size_t)t * DMODEL;
        for (int i = threadIdx.x; i < DMODEL; i += blockDim.x) dst[i] = f2bf(src[i]);
    }
}

// ---------------- bf16 WMMA GEMM: Out[e] = act(A[e] @ B[e] + bias[e]) ----------------
// A: bf16 [E,CAP,KDIM] row-major, staged into LDS via GLOBAL_LOAD_ASYNC_TO_LDS_B64 (ASYNCcnt).
// B: fp32 [E,KDIM,NDIM] row-major, converted to bf16 in-register, stored transposed in LDS.
// block tile 64x128, BK=32; 8 waves as 2x4; each wave 32x32 = 2x2 wmma_f32_16x16x32_bf16.
template<int KDIM, int NDIM, bool GELU, bool OUT_BF16>
__global__ __launch_bounds__(256) void k_gemm(const unsigned short* __restrict__ A,
                                              const float* __restrict__ B,
                                              const float* __restrict__ bias,
                                              void* __restrict__ Out) {
    __shared__ unsigned short As[64 * ASTRIDE];    // [row][k]
    __shared__ unsigned short Bs[128 * ASTRIDE];   // transposed [n][k]

    int tid = threadIdx.x;
    int lane = tid & 31, wid = tid >> 5;
    int wrow = wid >> 2, wcol = wid & 3;
    int e  = blockIdx.z;
    int m0 = blockIdx.y * 64;
    int n0 = blockIdx.x * 128;

    const unsigned short* Ae = A + ((size_t)e * CAP + m0) * KDIM;
    const float*          Be = B + (size_t)e * KDIM * NDIM + n0;

    // Per-thread async-copy slots for the A tile: 512 qwords (64 rows x 8 qwords), 2 per thread.
    // chunk c: row r = c>>3, qword q = c&7  (8B = 4 ushorts per qword)
    int c0 = tid, c1 = tid + 256;
    int r0 = c0 >> 3, q0 = c0 & 7;
    int r1 = c1 >> 3, q1 = c1 & 7;
    unsigned ldsA = (unsigned)(unsigned long long)
        (__attribute__((address_space(3))) unsigned short*)As;
    unsigned lds0 = ldsA + (unsigned)(r0 * ASTRIDE * 2 + q0 * 8);
    unsigned lds1 = ldsA + (unsigned)(r1 * ASTRIDE * 2 + q1 * 8);
    unsigned long long ga0 = (unsigned long long)(Ae + (size_t)r0 * KDIM + q0 * 4);
    unsigned long long ga1 = (unsigned long long)(Ae + (size_t)r1 * KDIM + q1 * 4);

    v8f c[2][2] = {};

    for (int kt = 0; kt < KDIM / 32; ++kt) {
        // --- A tile: pure bf16 copy, global -> LDS, async (no VGPR round-trip) ---
        unsigned long long g0 = ga0 + (unsigned long long)kt * 64;  // kt*32 ushorts
        unsigned long long g1 = ga1 + (unsigned long long)kt * 64;
        asm volatile("global_load_async_to_lds_b64 %0, %1, off"
                     :: "v"(lds0), "v"(g0) : "memory");
        asm volatile("global_load_async_to_lds_b64 %0, %1, off"
                     :: "v"(lds1), "v"(g1) : "memory");

        // --- B tile: 32x128 fp32 -> bf16, stored transposed [n][k]; float4 coalesced ---
        #pragma unroll
        for (int i = 0; i < 4; ++i) {
            int u = tid + i * 256;              // 1024 float4 chunks
            int kk = u >> 5, nn = (u & 31) * 4;
            float4 f4 = *(const float4*)(Be + (size_t)(kt * 32 + kk) * NDIM + nn);
            Bs[(nn + 0) * ASTRIDE + kk] = f2bf(f4.x);
            Bs[(nn + 1) * ASTRIDE + kk] = f2bf(f4.y);
            Bs[(nn + 2) * ASTRIDE + kk] = f2bf(f4.z);
            Bs[(nn + 3) * ASTRIDE + kk] = f2bf(f4.w);
        }
        if (kt + 1 < KDIM / 32) {
            __builtin_prefetch(Be + (size_t)(kt * 32 + 32) * NDIM + (tid & 127), 0, 0);
        }
        asm volatile("s_wait_asynccnt 0x0" ::: "memory");
        __syncthreads();

        // --- fragments per documented 16-bit A/B layouts ---
        int g  = lane >> 4;    // lane half
        int li = lane & 15;
        FragAB a[2], b[2];
        #pragma unroll
        for (int i = 0; i < 2; ++i) {
            const unsigned short* ap = &As[(wrow * 32 + 16 * i + li) * ASTRIDE];
            #pragma unroll
            for (int j = 0; j < 4; ++j) a[i].u[j]     = *(const unsigned*)(ap + g * 8 + 2 * j);
            #pragma unroll
            for (int j = 0; j < 4; ++j) a[i].u[4 + j] = *(const unsigned*)(ap + 16 + g * 8 + 2 * j);
        }
        #pragma unroll
        for (int j = 0; j < 2; ++j) {
            const unsigned short* bp = &Bs[(wcol * 32 + 16 * j + li) * ASTRIDE];
            #pragma unroll
            for (int q = 0; q < 4; ++q) b[j].u[q]     = *(const unsigned*)(bp + g * 8 + 2 * q);
            #pragma unroll
            for (int q = 0; q < 4; ++q) b[j].u[4 + q] = *(const unsigned*)(bp + 16 + g * 8 + 2 * q);
        }
        #pragma unroll
        for (int i = 0; i < 2; ++i)
            #pragma unroll
            for (int j = 0; j < 2; ++j)
                c[i][j] = __builtin_amdgcn_wmma_f32_16x16x32_bf16(
                    false, a[i].v, false, b[j].v, (short)0, c[i][j], false, false);
        __syncthreads();
    }

    // epilogue: C layout — VGPR r: M = r + 8*(lane/16), N = lane%16
    int g2 = lane >> 4, li = lane & 15;
    #pragma unroll
    for (int i = 0; i < 2; ++i)
        #pragma unroll
        for (int j = 0; j < 2; ++j) {
            int nn = n0 + wcol * 32 + 16 * j + li;
            float bv = bias[e * NDIM + nn];
            #pragma unroll
            for (int r = 0; r < 8; ++r) {
                int mm = m0 + wrow * 32 + 16 * i + r + 8 * g2;
                float val = c[i][j][r] + bv;
                if (GELU) val = 0.5f * val * (1.f + erff(val * 0.70710678118f));
                size_t off = ((size_t)e * CAP + mm) * NDIM + nn;
                if (OUT_BF16) ((unsigned short*)Out)[off] = f2bf(val);
                else          ((float*)Out)[off] = val;
            }
        }
}

// ---------------- combine: both routed slots per token known -> no atomics ----------------
__global__ void k_combine(const float* __restrict__ EO, const int* __restrict__ expert_idx,
                          const int* __restrict__ pair_pos, const float* __restrict__ gate,
                          float* __restrict__ y) {
    int t = blockIdx.x;
    int e0 = expert_idx[2 * t], e1 = expert_idx[2 * t + 1];
    int p0 = pair_pos[2 * t],  p1 = pair_pos[2 * t + 1];
    float g0 = gate[2 * t], g1 = gate[2 * t + 1];
    float wsum = g0 + g1;
    float inv = (wsum > 0.f) ? 1.f / wsum : 1.f;
    const float* r0 = EO + ((size_t)e0 * CAP + p0) * DMODEL;
    const float* r1 = EO + ((size_t)e1 * CAP + p1) * DMODEL;
    float* yr = y + (size_t)t * DMODEL;
    for (int d = threadIdx.x; d < DMODEL; d += blockDim.x)
        yr[d] = (r0[d] * g0 + r1[d] * g1) * inv;
}

// ---------------- load-balance loss ----------------
__global__ void k_loss(const float* imp, const float* loadc, float* out) {
    int lane = threadIdx.x;
    float v = imp[lane] * loadc[lane];
    for (int off = 16; off > 0; off >>= 1) v += __shfl_xor(v, off, 32);
    if (lane == 0) out[0] = v * (float)N_EXP / ((float)S_TOK * (float)S_TOK);
}

extern "C" void kernel_launch(void* const* d_in, const int* in_sizes, int n_in,
                              void* d_out, int out_size, void* d_ws, size_t ws_size,
                              hipStream_t stream) {
    const float* x  = (const float*)d_in[0];
    const float* wg = (const float*)d_in[1];
    const float* W1 = (const float*)d_in[2];
    const float* b1 = (const float*)d_in[3];
    const float* W2 = (const float*)d_in[4];
    const float* b2 = (const float*)d_in[5];
    float* out = (float*)d_out;

    char* ws = (char*)d_ws;
    size_t off = 0;
    auto alloc = [&](size_t bytes) -> void* {
        off = (off + 255) & ~(size_t)255;
        void* p = ws + off;
        off += bytes;
        return p;
    };
    int*            expert_idx = (int*)           alloc((size_t)S_TOK * 2 * sizeof(int));
    float*          gate       = (float*)         alloc((size_t)S_TOK * 2 * sizeof(float));
    int*            pair_pos   = (int*)           alloc((size_t)S_TOK * 2 * sizeof(int));
    int*            slot_pair  = (int*)           alloc((size_t)N_EXP * CAP * sizeof(int));
    float*          imp        = (float*)         alloc(N_EXP * sizeof(float));
    float*          loadc      = (float*)         alloc(N_EXP * sizeof(float));
    unsigned short* Xb         = (unsigned short*)alloc((size_t)N_EXP * CAP * DMODEL * 2); // 21 MB
    unsigned short* Hb         = (unsigned short*)alloc((size_t)N_EXP * CAP * DFF * 2);    // 84 MB
    float*          EO         = (float*)         alloc((size_t)N_EXP * CAP * DMODEL * 4); // 42 MB

    k_init<<<(N_EXP * CAP + 255) / 256, 256, 0, stream>>>(imp, loadc, slot_pair);
    k_router<<<S_TOK / 8, 256, 0, stream>>>(x, wg, expert_idx, gate, imp, loadc);
    k_scan<<<N_EXP, 32, 0, stream>>>(expert_idx, pair_pos, slot_pair);
    k_dispatch<<<N_EXP * CAP, 256, 0, stream>>>(x, slot_pair, Xb);
    k_gemm<DMODEL, DFF, true,  true ><<<dim3(DFF / 128,    CAP / 64, N_EXP), 256, 0, stream>>>(Xb, W1, b1, Hb);
    k_gemm<DFF,    DMODEL, false, false><<<dim3(DMODEL / 128, CAP / 64, N_EXP), 256, 0, stream>>>(Hb, W2, b2, EO);
    k_combine<<<S_TOK, 256, 0, stream>>>(EO, expert_idx, pair_pos, gate, out);
    k_loss<<<1, 32, 0, stream>>>(imp, loadc, out + (size_t)S_TOK * DMODEL);
}
